// RepCodec_12773232738561
// MI455X (gfx1250) — matrix operand order repaired and checked by reading, not compiled, not run on hardware
//
#include <hip/hip_runtime.h>
#include <hip/hip_bf16.h>

#define CC 768
#define TT 4096
#define KB 1024
#define BB 8

typedef __attribute__((ext_vector_type(16))) _Float16 v16h;
typedef __attribute__((ext_vector_type(8)))  _Float16 v8h;
typedef __attribute__((ext_vector_type(8)))  float    v8f;

union F16x16 { v16h v; v8h h[2]; };

// ---------------------------------------------------------------------------
// Prep kernels
// ---------------------------------------------------------------------------
// x f32 [B][C][T] -> x16 f16 [B][T][C]
__global__ __launch_bounds__(256) void cvt_x_tr(const float* __restrict__ src,
                                                _Float16* __restrict__ dst) {
    size_t gid = (size_t)blockIdx.x * 256 + threadIdx.x;
    const size_t S = (size_t)BB * CC * TT;
    if (gid < S) {
        int t = (int)(gid % TT);
        size_t r = gid / TT;
        int c = (int)(r % CC);
        int b = (int)(r / CC);
        dst[((size_t)b * TT + t) * CC + c] = (_Float16)src[gid];
    }
}

// src: [S][C][C][k] (Cout,Cin,k) -> dst: [S][C][k][C] (Cout,k,Cin) f16
__global__ __launch_bounds__(256) void cvt_weight(const float* __restrict__ src,
                                                  _Float16* __restrict__ dst,
                                                  int S, int k) {
    size_t n = (size_t)S * CC * CC * k;
    size_t gid = (size_t)blockIdx.x * 256 + threadIdx.x;
    if (gid >= n) return;
    int dk = (int)(gid % k);
    size_t r = gid / k;
    int ci = (int)(r % CC);
    size_t r2 = r / CC;
    int co = (int)(r2 % CC);
    int s  = (int)(r2 / CC);
    dst[(((size_t)s * CC + co) * k + dk) * CC + ci] = (_Float16)src[gid];
}

// embed f32 [C][K] -> eT16 f16 [K][C]
__global__ __launch_bounds__(256) void cvt_embed(const float* __restrict__ src,
                                                 _Float16* __restrict__ dst) {
    int i = blockIdx.x * 256 + threadIdx.x;
    if (i < CC * KB) {
        int c = i / KB, kk = i % KB;
        dst[(size_t)kk * CC + c] = (_Float16)src[i];
    }
}

__global__ __launch_bounds__(256) void calc_enorm(const float* __restrict__ src,
                                                  float* __restrict__ enorm) {
    int kk = blockIdx.x * 256 + threadIdx.x;
    if (kk < KB) {
        float s = 0.f;
        for (int c = 0; c < CC; ++c) { float v = src[(size_t)c * KB + kk]; s += v * v; }
        enorm[kk] = s;
    }
}

__global__ __launch_bounds__(256) void zero_f32(float* __restrict__ p, int n) {
    int i = blockIdx.x * 256 + threadIdx.x;
    if (i < n) p[i] = 0.f;
}

// ---------------------------------------------------------------------------
// Implicit-GEMM conv1d ('same', stride 1) via WMMA f16->f32, time-major data.
//   in16  [B][T][C], wB [Cout][K][Cin] (f16), out [B][T][C]
//   GEMM tile: M=16 time steps (A = x rows, contiguous in C),
//              N=16 out-channels (B = weights, per-lane contiguous in Cin),
//              K-loop over Cin in steps of 32, unrolled over the K taps.
//   Epilogue: +bias, +residual, raw f16 / elu f16 / f32 ([B][C][T]) outputs.
// No LDS, no barriers: every fragment load is a contiguous global b128.
// ---------------------------------------------------------------------------
template <int K>
__global__ __launch_bounds__(256) void conv_wmma(
    const _Float16* __restrict__ in16, const _Float16* __restrict__ wB,
    const float* __restrict__ bias, const _Float16* __restrict__ resid,
    _Float16* __restrict__ out16, _Float16* __restrict__ oelu16,
    float* __restrict__ outf32)
{
    const int tid  = threadIdx.x;
    const int b    = blockIdx.z;
    const int t0   = blockIdx.x * 16;
    const int wave = tid >> 5;
    const int lane = tid & 31;
    const int nloc = lane & 15;
    const int hi   = lane >> 4;
    const int co_base = (blockIdx.y * 8 + wave) * 16;
    const int pad = (K - 1) >> 1;
    const size_t base = (size_t)b * TT * CC;
    const v8h zv = {};
    v8f acc = {};

#pragma unroll
    for (int dk = 0; dk < K; ++dk) {
        const int tm = t0 + nloc + dk - pad;          // A row for this lane
        const bool ok = (tm >= 0) && (tm < TT);       // 'same' zero padding
        const _Float16* xrow = in16 + base + (size_t)tm * CC;
        const _Float16* wrow = wB + ((size_t)(co_base + nloc) * K + dk) * CC;
        for (int ci0 = 0; ci0 < CC; ci0 += 32) {
            F16x16 au, bu;
            if (ok) {
                au.h[0] = *(const v8h*)(xrow + ci0 + hi * 8);
                au.h[1] = *(const v8h*)(xrow + ci0 + 16 + hi * 8);
            } else {
                au.h[0] = zv; au.h[1] = zv;
            }
            bu.h[0] = *(const v8h*)(wrow + ci0 + hi * 8);
            bu.h[1] = *(const v8h*)(wrow + ci0 + 16 + hi * 8);
            acc = __builtin_amdgcn_wmma_f32_16x16x32_f16(
                false, au.v, false, bu.v, (short)0, acc, false, false);
        }
    }

    const int co = co_base + nloc;
    const float bv = bias ? bias[co] : 0.f;
#pragma unroll
    for (int r = 0; r < 8; ++r) {
        const int t = t0 + hi * 8 + r;
        const size_t off = base + (size_t)t * CC + co;
        float v = acc[r] + bv;
        if (resid)  v += (float)resid[off];
        if (out16)  out16[off] = (_Float16)v;
        if (oelu16) oelu16[off] = (_Float16)(v > 0.f ? v : __expf(v) - 1.f);
        if (outf32) outf32[(size_t)b * CC * TT + (size_t)co * TT + t] = v;
    }
}

// ---------------------------------------------------------------------------
// VQ: s[n,k] = z_n . e_k via WMMA; argmin dist == argmax(2s - |e|^2).
// z16 [B*T][C] time-major -> A rows contiguous; eT16 [K][C] -> B per-lane
// contiguous. Each wave owns one 16-position tile, loops 64 codebook tiles.
// ---------------------------------------------------------------------------
__global__ __launch_bounds__(256) void vq_argmin(
    const _Float16* __restrict__ z16, const _Float16* __restrict__ eT16,
    const float* __restrict__ enorm, int* __restrict__ idxbuf,
    float* __restrict__ counts)
{
    const int tid  = threadIdx.x;
    const int wave = tid >> 5;
    const int lane = tid & 31;
    const int nloc = lane & 15;
    const int hi   = lane >> 4;
    const int pos0 = (blockIdx.x * 8 + wave) * 16;
    const _Float16* arow = z16 + (size_t)(pos0 + nloc) * CC;

    float best[8];
    int   bidx[8];
#pragma unroll
    for (int r = 0; r < 8; ++r) { best[r] = -1e30f; bidx[r] = 0; }

    for (int k0 = 0; k0 < KB; k0 += 16) {
        v8f acc = {};
        const _Float16* erow = eT16 + (size_t)(k0 + nloc) * CC;
        for (int ci0 = 0; ci0 < CC; ci0 += 32) {
            F16x16 au, bu;
            au.h[0] = *(const v8h*)(arow + ci0 + hi * 8);
            au.h[1] = *(const v8h*)(arow + ci0 + 16 + hi * 8);
            bu.h[0] = *(const v8h*)(erow + ci0 + hi * 8);
            bu.h[1] = *(const v8h*)(erow + ci0 + 16 + hi * 8);
            acc = __builtin_amdgcn_wmma_f32_16x16x32_f16(
                false, au.v, false, bu.v, (short)0, acc, false, false);
        }
        const float en = enorm[k0 + nloc];
#pragma unroll
        for (int r = 0; r < 8; ++r) {
            float sc = 2.f * acc[r] - en;
            if (sc > best[r]) { best[r] = sc; bidx[r] = k0 + nloc; }
        }
    }
    // argmax across the 16 lanes of each half (tie -> lower index, matches argmin-first)
    for (int off = 1; off < 16; off <<= 1) {
#pragma unroll
        for (int r = 0; r < 8; ++r) {
            float ov = __shfl_xor(best[r], off, 32);
            int   oi = __shfl_xor(bidx[r], off, 32);
            if (ov > best[r] || (ov == best[r] && oi < bidx[r])) { best[r] = ov; bidx[r] = oi; }
        }
    }
    if (nloc == 0) {
#pragma unroll
        for (int r = 0; r < 8; ++r) {
            int pos = pos0 + hi * 8 + r;
            idxbuf[pos] = bidx[r];
            atomicAdd(&counts[bidx[r]], 1.0f);
        }
    }
}

// zq = e[:, idx] (straight-through forward value); commitment-loss partials.
// Writes zq f32 in [B][C][T] (d_out) and zq f16 in [B][T][C] (decoder input).
__global__ __launch_bounds__(256) void dequant(
    const int* __restrict__ idxbuf, const float* __restrict__ embed,
    const float* __restrict__ zf32, float* __restrict__ zqf32,
    _Float16* __restrict__ zq16, float* __restrict__ lossAcc)
{
    size_t gid = (size_t)blockIdx.x * 256 + threadIdx.x;
    const size_t S = (size_t)BB * CC * TT;
    float d2 = 0.f;
    if (gid < S) {
        int t = (int)(gid % TT);
        size_t r = gid / TT;
        int c = (int)(r % CC);
        int b = (int)(r / CC);
        int id = idxbuf[b * TT + t];
        float q = embed[(size_t)c * KB + id];
        float d = q - zf32[gid];
        d2 = d * d;
        zqf32[gid] = q;
        zq16[((size_t)b * TT + t) * CC + c] = (_Float16)q;
    }
    for (int off = 16; off > 0; off >>= 1) d2 += __shfl_xor(d2, off, 32);
    if ((threadIdx.x & 31) == 0) atomicAdd(lossAcc, d2);
}

__global__ void finalize_vq(const float* __restrict__ lossAcc,
                            const float* __restrict__ counts,
                            float* __restrict__ out_loss,
                            float* __restrict__ out_perp)
{
    if (threadIdx.x == 0 && blockIdx.x == 0) {
        const float N = (float)(BB * TT);
        *out_loss = *lossAcc / (N * (float)CC);
        float s = 0.f;
        for (int kk = 0; kk < KB; ++kk) {
            float avg = counts[kk] / N;
            s += avg * __logf(avg + 1e-10f);
        }
        *out_perp = __expf(-s);
    }
}

// ---------------------------------------------------------------------------
extern "C" void kernel_launch(void* const* d_in, const int* in_sizes, int n_in,
                              void* d_out, int out_size, void* d_ws, size_t ws_size,
                              hipStream_t stream) {
    (void)in_sizes; (void)n_in; (void)out_size; (void)ws_size;

    const float* x           = (const float*)d_in[0];
    const float* enc_conv0_w = (const float*)d_in[1];
    const float* enc_ru_w1   = (const float*)d_in[2];
    const float* enc_ru_b1   = (const float*)d_in[3];
    const float* enc_ru_w2   = (const float*)d_in[4];
    const float* enc_ru_b2   = (const float*)d_in[5];
    const float* enc_blk_w   = (const float*)d_in[6];
    const float* enc_blk_b   = (const float*)d_in[7];
    const float* proj_w      = (const float*)d_in[8];
    const float* vq_embed    = (const float*)d_in[9];
    const float* dec_conv1_w = (const float*)d_in[10];
    const float* dec_blk_w   = (const float*)d_in[11];
    const float* dec_blk_b   = (const float*)d_in[12];
    const float* dec_ru_w1   = (const float*)d_in[13];
    const float* dec_ru_b1   = (const float*)d_in[14];
    const float* dec_ru_w2   = (const float*)d_in[15];
    const float* dec_ru_b2   = (const float*)d_in[16];
    const float* dec_conv2_w = (const float*)d_in[17];

    const size_t S = (size_t)BB * CC * TT;  // 25165824
    float* d_y    = (float*)d_out;
    float* d_zq   = d_y + S;
    float* d_z    = d_zq + S;
    float* d_loss = d_z + S;
    float* d_perp = d_loss + 1;

    char* p = (char*)d_ws;
    auto alloc = [&](size_t bytes) -> char* {
        char* r = p; p += (bytes + 255) & ~(size_t)255; return r;
    };
    _Float16* bufs[4];
    for (int i = 0; i < 4; ++i) bufs[i] = (_Float16*)alloc(S * 2);
    const size_t W3 = (size_t)CC * CC * 3, W1 = (size_t)CC * CC;
    _Float16* w_enc0 = (_Float16*)alloc(W3 * 2);
    _Float16* w_eru1 = (_Float16*)alloc(4 * W3 * 2);
    _Float16* w_eru2 = (_Float16*)alloc(4 * W1 * 2);
    _Float16* w_eblk = (_Float16*)alloc(2 * W3 * 2);
    _Float16* w_proj = (_Float16*)alloc(W3 * 2);
    _Float16* w_dec1 = (_Float16*)alloc(W3 * 2);
    _Float16* w_dblk = (_Float16*)alloc(2 * W3 * 2);
    _Float16* w_dru1 = (_Float16*)alloc(4 * W3 * 2);
    _Float16* w_dru2 = (_Float16*)alloc(4 * W1 * 2);
    _Float16* w_dec2 = (_Float16*)alloc(W3 * 2);
    _Float16* eT16   = (_Float16*)alloc((size_t)KB * CC * 2);
    float* enorm     = (float*)alloc(KB * 4);
    float* counts    = (float*)alloc(KB * 4);
    float* lossAcc   = (float*)alloc(4);   // contiguous after counts
    int*   idxbuf    = (int*)alloc((size_t)BB * TT * 4);

    auto blks = [](size_t n) { return (unsigned)((n + 255) / 256); };

    // --- prep ---
    cvt_weight<<<blks(W3),     256, 0, stream>>>(enc_conv0_w, w_enc0, 1, 3);
    cvt_weight<<<blks(4 * W3), 256, 0, stream>>>(enc_ru_w1,   w_eru1, 4, 3);
    cvt_weight<<<blks(4 * W1), 256, 0, stream>>>(enc_ru_w2,   w_eru2, 4, 1);
    cvt_weight<<<blks(2 * W3), 256, 0, stream>>>(enc_blk_w,   w_eblk, 2, 3);
    cvt_weight<<<blks(W3),     256, 0, stream>>>(proj_w,      w_proj, 1, 3);
    cvt_weight<<<blks(W3),     256, 0, stream>>>(dec_conv1_w, w_dec1, 1, 3);
    cvt_weight<<<blks(2 * W3), 256, 0, stream>>>(dec_blk_w,   w_dblk, 2, 3);
    cvt_weight<<<blks(4 * W3), 256, 0, stream>>>(dec_ru_w1,   w_dru1, 4, 3);
    cvt_weight<<<blks(4 * W1), 256, 0, stream>>>(dec_ru_w2,   w_dru2, 4, 1);
    cvt_weight<<<blks(W3),     256, 0, stream>>>(dec_conv2_w, w_dec2, 1, 3);
    cvt_embed<<<blks((size_t)CC * KB), 256, 0, stream>>>(vq_embed, eT16);
    calc_enorm<<<blks(KB), 256, 0, stream>>>(vq_embed, enorm);
    cvt_x_tr<<<blks(S), 256, 0, stream>>>(x, bufs[0]);
    zero_f32<<<blks(KB + 1), 256, 0, stream>>>(counts, KB + 1);

    dim3 cg(TT / 16, CC / 128, BB);
    auto conv = [&](const _Float16* in, const _Float16* w, const float* bias,
                    const _Float16* res, _Float16* o16, _Float16* oelu,
                    float* o32, int k) {
        if (k == 3) conv_wmma<3><<<cg, 256, 0, stream>>>(in, w, bias, res, o16, oelu, o32);
        else        conv_wmma<1><<<cg, 256, 0, stream>>>(in, w, bias, res, o16, oelu, o32);
    };
    auto pick = [](int a, int b, int c) {
        for (int i = 0; i < 4; ++i) if (i != a && i != b && i != c) return i;
        return 0;
    };

    int raw = 0, elu = -1;
    // ResidualUnit: y1 = conv3(elu(h))+b1 ; h' = h + conv1(elu(y1))+b2
    auto ru = [&](const _Float16* w1, const float* b1, const _Float16* w2,
                  const float* b2, bool need_elu) {
        int y1 = pick(raw, elu, -1);
        conv(bufs[elu], w1, b1, nullptr, nullptr, bufs[y1], nullptr, 3);
        int nraw = pick(raw, y1, -1);
        int nelu = pick(raw, y1, nraw);
        conv(bufs[y1], w2, b2, bufs[raw], bufs[nraw],
             need_elu ? bufs[nelu] : nullptr, nullptr, 1);
        raw = nraw; elu = need_elu ? nelu : -1;
    };
    auto pconv = [&](const _Float16* w, const float* bias, bool need_elu,
                     float* o32, int k) {
        int nraw = pick(raw, -1, -1);
        int nelu = need_elu ? pick(raw, nraw, -1) : -1;
        conv(bufs[raw], w, bias, nullptr, bufs[nraw],
             need_elu ? bufs[nelu] : nullptr, o32, k);
        raw = nraw; elu = nelu;
    };

    // --- encoder ---
    pconv(w_enc0, nullptr, true, nullptr, 3);
    for (int i = 0; i < 2; ++i) {
        for (int j = 0; j < 2; ++j) {
            size_t u = (size_t)i * 2 + j;
            ru(w_eru1 + u * W3, enc_ru_b1 + u * CC,
               w_eru2 + u * W1, enc_ru_b2 + u * CC, /*need_elu=*/j == 0);
        }
        pconv(w_eblk + (size_t)i * W3, enc_blk_b + (size_t)i * CC,
              /*need_elu=*/i == 0, nullptr, 3);
    }

    // --- projector: z (f16 for VQ, f32 to d_out) ---
    int iz = pick(raw, -1, -1);
    conv(bufs[raw], w_proj, nullptr, nullptr, bufs[iz], nullptr, d_z, 3);

    // --- quantizer ---
    vq_argmin<<<dim3(BB * TT / 16 / 8), 256, 0, stream>>>(bufs[iz], eT16, enorm,
                                                          idxbuf, counts);
    int iq = pick(iz, -1, -1);
    dequant<<<blks(S), 256, 0, stream>>>(idxbuf, vq_embed, d_z, d_zq, bufs[iq], lossAcc);
    finalize_vq<<<1, 32, 0, stream>>>(lossAcc, counts, d_loss, d_perp);

    // --- decoder ---
    raw = iq; elu = -1;
    pconv(w_dec1, nullptr, false, nullptr, 3);
    for (int i = 0; i < 2; ++i) {
        pconv(w_dblk + (size_t)i * W3, dec_blk_b + (size_t)i * CC, true, nullptr, 3);
        for (int j = 0; j < 2; ++j) {
            size_t u = (size_t)i * 2 + j;
            ru(w_dru1 + u * W3, dec_ru_b1 + u * CC,
               w_dru2 + u * W1, dec_ru_b2 + u * CC, /*need_elu=*/j == 0);
        }
    }
    conv(bufs[raw], w_dec2, nullptr, nullptr, nullptr, nullptr, d_y, 3);
}
